// DAGAM_24206435680718
// MI455X (gfx1250) — matrix operand order repaired and checked by reading, not compiled, not run on hardware
//
#include <hip/hip_runtime.h>

typedef __attribute__((ext_vector_type(16))) __bf16 v16bf;
typedef __attribute__((ext_vector_type(8)))  float  v8f;
typedef __attribute__((ext_vector_type(4)))  int    i32x4;

#define B_SZ    16
#define NFULL   8192
#define MHALF   4096
#define C_DIM   128
#define QKV_N   384

#define QKVP_ELEMS  (24*4*32*16)   // 24 col-tiles x 4 k-tiles x 32 lanes x 16 bf16
#define PROJP_ELEMS (8*4*32*16)    // 8 col-tiles x 4 k-tiles
#define WF_ELEMS_PER (8*4*32*16)   // one 128x128 packed B matrix

// ---------------------------------------------------------------------------
// Async global->LDS copy (CDNA5 GLOBAL_LOAD_ASYNC_TO_LDS_B128, ASYNCcnt
// tracked). Builtin takes (v4i AS(1)*, v4i AS(3)*, imm offset, imm cpol).
// Falls back to a plain 16B copy if the builtin is unavailable.
// ---------------------------------------------------------------------------
#if defined(__gfx1250__) && __has_builtin(__builtin_amdgcn_global_load_async_to_lds_b128)
#define USE_ASYNC_LDS 1
#else
#define USE_ASYNC_LDS 0
#endif

__device__ __forceinline__ void async_copy16(const void* g, void* l)
{
#if USE_ASYNC_LDS
    __builtin_amdgcn_global_load_async_to_lds_b128(
        (__attribute__((address_space(1))) i32x4*)g,
        (__attribute__((address_space(3))) i32x4*)l,
        0, 0);
#else
    *(uint4*)l = *(const uint4*)g;
#endif
}

__device__ __forceinline__ void async_wait_all()
{
#if USE_ASYNC_LDS
#if __has_builtin(__builtin_amdgcn_s_wait_asynccnt)
    __builtin_amdgcn_s_wait_asynccnt(0);
#else
    asm volatile("s_wait_asynccnt 0" ::: "memory");
#endif
#endif
}

// ---------------------------------------------------------------------------
// Kernel 0: pack qkv_w (128x384) and proj_w (128x128) into bf16 WMMA
// B-fragment layout: [(ct*KT + kt)*32 + lane]*16 + e ; k = kt*32 + e + 16*(lane>>4)
// col = ct*16 + (lane&15)
// ---------------------------------------------------------------------------
__global__ __launch_bounds__(256) void pack_weights_kernel(
    const float* __restrict__ qkv_w, const float* __restrict__ proj_w,
    __bf16* __restrict__ qkvp, __bf16* __restrict__ projp)
{
    int idx = blockIdx.x * 256 + threadIdx.x;
    if (idx < QKVP_ELEMS) {
        int e = idx & 15, lane = (idx >> 4) & 31, kt = (idx >> 9) & 3, ct = idx >> 11;
        int col = ct*16 + (lane & 15);
        int k   = kt*32 + e + 16*(lane >> 4);
        qkvp[idx] = (__bf16)qkv_w[k*QKV_N + col];
    }
    int j = idx - QKVP_ELEMS;
    if (j >= 0 && j < PROJP_ELEMS) {
        int e = j & 15, lane = (j >> 4) & 31, kt = (j >> 9) & 3, ct = j >> 11;
        int col = ct*16 + (lane & 15);
        int k   = kt*32 + e + 16*(lane >> 4);
        projp[j] = (__bf16)proj_w[k*C_DIM + col];
    }
}

// ---------------------------------------------------------------------------
// Kernel 1: per (batch, 64-position block): QKV via WMMA bf16, per-position
// 8x8 head attention via VALU, write attn/attn1/attn2 bf16 + per-block
// deterministic channel sum/max partials.
// LDS: xa32 32K + xb32 32K + t1 48K + t2 32K + staging 48K = 192KB (1 WG/WGP)
// ---------------------------------------------------------------------------
__global__ __launch_bounds__(128) void qkv_headattn_kernel(
    const float* __restrict__ x,
    const __bf16* __restrict__ qkvp,
    __bf16* __restrict__ attn_d, __bf16* __restrict__ attn1_d,
    __bf16* __restrict__ attn2_d,
    float* __restrict__ psum, float* __restrict__ pmax)
{
    extern __shared__ char smem[];
    float*  xaf = (float*)smem;            // 64 x 128 f32 (async-loaded)
    float*  xbf = xaf + 64*128;            // 64 x 128 f32
    __bf16* t1  = (__bf16*)(xbf + 64*128); // 64 x 384  (q|k1|v1)
    __bf16* t2  = t1 + 64*384;             // 64 x 256  (k2|v2)
    __bf16* st  = t2 + 64*256;             // 3 x 64 x 128 staging

    const int tid  = threadIdx.x;
    const int b    = blockIdx.x >> 6;
    const int blk  = blockIdx.x & 63;
    const int tile0 = blk * 64;

    // Phase 1: async bulk copy of the two contiguous 32KB x tiles into LDS
    const float* x1p = x + ((size_t)b*NFULL + tile0) * C_DIM;
    const float* x2p = x + ((size_t)b*NFULL + MHALF + tile0) * C_DIM;
    #pragma unroll
    for (int i = 0; i < 16; ++i) {
        int lin = tid + i*128;                 // 0..2047 16-byte slots
        async_copy16(x1p + lin*4, xaf + lin*4);
        async_copy16(x2p + lin*4, xbf + lin*4);
    }
    async_wait_all();
    __syncthreads();

    // Phase 2: WMMA QKV, each wave owns 16 rows
    const int w = tid >> 5, lane = tid & 31;
    const int rr = lane & 15, hh = lane >> 4;
    const int r0 = w * 16;

    v16bf af[4];
    #pragma unroll
    for (int kt = 0; kt < 4; ++kt) {
        #pragma unroll
        for (int e = 0; e < 16; ++e) {
            int k = kt*32 + e + 16*hh;
            af[kt][e] = (__bf16)xaf[(r0 + rr)*C_DIM + k];
        }
    }
    for (int ct = 0; ct < 24; ++ct) {             // full q|k|v for x1
        v8f acc = {};
        #pragma unroll
        for (int kt = 0; kt < 4; ++kt) {
            v16bf bfr = *(const v16bf*)(qkvp + ((size_t)(ct*4 + kt)*32 + lane)*16);
            acc = __builtin_amdgcn_wmma_f32_16x16x32_bf16(false, af[kt], false, bfr,
                                                          (short)0, acc, false, false);
        }
        #pragma unroll
        for (int i = 0; i < 8; ++i)
            t1[(r0 + i + 8*hh)*QKV_N + ct*16 + rr] = (__bf16)acc[i];
    }
    #pragma unroll
    for (int kt = 0; kt < 4; ++kt) {
        #pragma unroll
        for (int e = 0; e < 16; ++e) {
            int k = kt*32 + e + 16*hh;
            af[kt][e] = (__bf16)xbf[(r0 + rr)*C_DIM + k];
        }
    }
    for (int ct = 8; ct < 24; ++ct) {             // only k|v for x2
        v8f acc = {};
        #pragma unroll
        for (int kt = 0; kt < 4; ++kt) {
            v16bf bfr = *(const v16bf*)(qkvp + ((size_t)(ct*4 + kt)*32 + lane)*16);
            acc = __builtin_amdgcn_wmma_f32_16x16x32_bf16(false, af[kt], false, bfr,
                                                          (short)0, acc, false, false);
        }
        #pragma unroll
        for (int i = 0; i < 8; ++i)
            t2[(r0 + i + 8*hh)*256 + (ct - 8)*16 + rr] = (__bf16)acc[i];
    }
    __syncthreads();

    // Phase 3: per-position 8x8 head attention (thread = (position, 4 heads))
    {
        const int p  = tid >> 1;
        const int h0 = (tid & 1) * 4;
        const __bf16* row1 = t1 + p*QKV_N;
        const __bf16* row2 = t2 + p*256;
        #pragma unroll
        for (int hi = 0; hi < 4; ++hi) {
            int h = h0 + hi;
            float q[16];
            #pragma unroll
            for (int d = 0; d < 16; ++d) q[d] = (float)row1[d*8 + h];
            float s1[8], s2[8];
            #pragma unroll
            for (int g = 0; g < 8; ++g) {
                float a1 = 0.f, a2 = 0.f;
                #pragma unroll
                for (int d = 0; d < 16; ++d) {
                    a1 += q[d] * (float)row1[128 + d*8 + g];   // k1
                    a2 += q[d] * (float)row2[      d*8 + g];   // k2
                }
                s1[g] = a1; s2[g] = a2;
            }
            float m1 = s1[0], m2 = s2[0];
            #pragma unroll
            for (int g = 1; g < 8; ++g) { m1 = fmaxf(m1, s1[g]); m2 = fmaxf(m2, s2[g]); }
            float w1[8], w2[8], Z1 = 0.f, Z2 = 0.f;
            #pragma unroll
            for (int g = 0; g < 8; ++g) {
                w1[g] = __expf(s1[g] - m1); Z1 += w1[g];
                w2[g] = __expf(s2[g] - m2); Z2 += w2[g];
            }
            float iZ1 = 1.f/Z1, iZ2 = 1.f/Z2;
            #pragma unroll
            for (int d = 0; d < 16; ++d) {
                float A1 = 0.f, A2 = 0.f;
                #pragma unroll
                for (int g = 0; g < 8; ++g) {
                    A1 += w1[g] * (float)row1[256 + d*8 + g];  // v1
                    A2 += w2[g] * (float)row2[128 + d*8 + g];  // v2
                }
                A1 *= iZ1; A2 *= iZ2;
                int c = d*8 + h;                               // cm channel layout
                st[0*8192 + p*C_DIM + c] = (__bf16)(A1 - A2);
                st[1*8192 + p*C_DIM + c] = (__bf16)A1;
                st[2*8192 + p*C_DIM + c] = (__bf16)A2;
            }
        }
    }
    __syncthreads();

    // Phase 4: coalesced writeback + deterministic per-block channel stats
    {
        size_t rowoff = ((size_t)b*MHALF + tile0) * C_DIM;
        const uint4* s0  = (const uint4*)(st);
        const uint4* s1v = (const uint4*)(st + 8192);
        const uint4* s2v = (const uint4*)(st + 16384);
        uint4* d0 = (uint4*)(attn_d  + rowoff);
        uint4* d1 = (uint4*)(attn1_d + rowoff);
        uint4* d2 = (uint4*)(attn2_d + rowoff);
        #pragma unroll
        for (int i = 0; i < 8; ++i) {
            int lin = tid + i*128;
            d0[lin] = s0[lin]; d1[lin] = s1v[lin]; d2[lin] = s2v[lin];
        }
    }
    {
        int c = tid;   // 128 threads own 128 channels
        #pragma unroll
        for (int mat = 0; mat < 3; ++mat) {
            const __bf16* s = st + mat*8192;
            float sum = 0.f, mx = -3.0e38f;
            for (int p = 0; p < 64; ++p) {
                float v = (float)s[p*C_DIM + c];
                sum += v; mx = fmaxf(mx, v);
            }
            size_t o = ((size_t)(mat*B_SZ + b)*64 + blk)*C_DIM + c;
            psum[o] = sum; pmax[o] = mx;
        }
    }
}

// ---------------------------------------------------------------------------
// Kernel 2: per batch: channel attention (ch, ch1, ch2), column-softmax
// (exploiting ch>0 => colmax[d] = max(ch)*ch1[d]), then Wf = softmax(A)@proj_w
// via WMMA, stored directly in packed B-fragment layout for kernel 3.
// ---------------------------------------------------------------------------
__global__ __launch_bounds__(128) void channel_mix_kernel(
    const float* __restrict__ psum, const float* __restrict__ pmax,
    const float* __restrict__ ca_w1, const float* __restrict__ ca_w2,
    const __bf16* __restrict__ projp, __bf16* __restrict__ wfp)
{
    extern __shared__ char smem[];
    float* sm_ch  = (float*)smem;        // 3*128 (ch, ch1, ch2)
    float* sm_v   = sm_ch + 3*128;       // 2*128 (avg, max)
    float* sm_hid = sm_v + 2*128;        // 64 hidden
    float* sm_iZ  = sm_hid + 64;         // 128
    __bf16* G     = (__bf16*)(sm_iZ + 128);   // 128x128 softmax matrix (bf16)

    const int tid = threadIdx.x;
    const int b   = blockIdx.x;

    for (int mat = 0; mat < 3; ++mat) {
        float s = 0.f, mx = -3.0e38f;
        const float* ps = psum + (size_t)(mat*B_SZ + b)*64*C_DIM + tid;
        const float* pm = pmax + (size_t)(mat*B_SZ + b)*64*C_DIM + tid;
        for (int blk = 0; blk < 64; ++blk) {
            s  += ps[blk*C_DIM];
            mx  = fmaxf(mx, pm[blk*C_DIM]);
        }
        sm_v[tid]       = s * (1.f/(float)MHALF);
        sm_v[128 + tid] = mx;
        __syncthreads();
        if (tid < 64) {                         // 32 hidden x {avg,max}
            int jj = tid & 31, ss = tid >> 5;
            const float* v    = sm_v + ss*128;
            const float* wrow = ca_w1 + jj*C_DIM;
            float hsum = 0.f;
            for (int c = 0; c < 128; ++c) hsum += v[c] * wrow[c];
            sm_hid[tid] = fmaxf(hsum, 0.f);     // relu
        }
        __syncthreads();
        float o = 0.f;
        const float* w2row = ca_w2 + tid*32;
        for (int j = 0; j < 32; ++j) o += (sm_hid[j] + sm_hid[32 + j]) * w2row[j];
        sm_ch[mat*128 + tid] = 1.f/(1.f + __expf(-o));   // sigmoid
        __syncthreads();
    }

    float chmax = 0.f;                          // sigmoid > 0
    for (int c = 0; c < 128; ++c) chmax = fmaxf(chmax, sm_ch[c]);

    const int w = tid >> 5, lane = tid & 31, rr = lane & 15, hh = lane >> 4;

    for (int pass = 0; pass < 2; ++pass) {      // pass0: A (ch1), pass1: AA (ch2)
        const float* chp = sm_ch + (1 + pass)*128;
        {   // 1/Z[d]
            float t = chp[tid], Z = 0.f;
            for (int c = 0; c < 128; ++c) Z += __expf((sm_ch[c] - chmax)*t);
            sm_iZ[tid] = 1.f/Z;
        }
        __syncthreads();
        {   // G[c][d] = softmax over c
            float chc = sm_ch[tid] - chmax;
            for (int d = 0; d < 128; ++d)
                G[tid*128 + d] = (__bf16)(__expf(chc * chp[d]) * sm_iZ[d]);
        }
        __syncthreads();
        // Wf = G @ proj_w, 8x8 tiles, write packed B-fragment layout
        __bf16* dst = wfp + (size_t)(pass*B_SZ + b)*WF_ELEMS_PER;
        for (int t = 0; t < 16; ++t) {
            int tile = w*16 + t, ti = tile >> 3, tj = tile & 7;
            v8f acc = {};
            #pragma unroll
            for (int kt = 0; kt < 4; ++kt) {
                v16bf a;
                #pragma unroll
                for (int e = 0; e < 16; ++e) {
                    int k = kt*32 + e + 16*hh;
                    a[e] = G[(ti*16 + rr)*128 + k];
                }
                v16bf bfr = *(const v16bf*)(projp + ((size_t)(tj*4 + kt)*32 + lane)*16);
                acc = __builtin_amdgcn_wmma_f32_16x16x32_bf16(false, a, false, bfr,
                                                              (short)0, acc, false, false);
            }
            #pragma unroll
            for (int i = 0; i < 8; ++i) {       // D(m,n) -> packed B(k=c, col=e)
                int cc = ti*16 + i + 8*hh;
                int lanep = rr + 16*((cc >> 4) & 1);
                dst[((size_t)(tj*4 + (cc >> 5))*32 + lanep)*16 + (cc & 15)] = (__bf16)acc[i];
            }
        }
        __syncthreads();
    }
}

// ---------------------------------------------------------------------------
// Kernel 3: out[b,n,:] = attn[b,m,:]@Wf[half,b] + attn_i[b,m,:]@proj_w + bias
// ---------------------------------------------------------------------------
__global__ __launch_bounds__(128) void fused_proj_kernel(
    const __bf16* __restrict__ attn_d, const __bf16* __restrict__ attn1_d,
    const __bf16* __restrict__ attn2_d,
    const __bf16* __restrict__ wfp, const __bf16* __restrict__ projp,
    const float* __restrict__ proj_b, float* __restrict__ out)
{
    extern __shared__ char smem[];
    __bf16* ya = (__bf16*)smem;    // 64 x 128 attn rows
    __bf16* yb = ya + 64*128;      // 64 x 128 attn1/attn2 rows

    const int tid = threadIdx.x;
    const int b   = blockIdx.x >> 7;
    const int t0  = (blockIdx.x & 127) * 64;
    const int half = (t0 >= MHALF) ? 1 : 0;
    const int m0  = t0 - half*MHALF;

    const __bf16* wf = wfp + (size_t)(half*B_SZ + b)*WF_ELEMS_PER;
    __builtin_prefetch(wf + (size_t)tid*128, 0, 1);
    __builtin_prefetch(projp + (size_t)tid*128, 0, 1);

    const __bf16* src_a = attn_d + ((size_t)b*MHALF + m0)*C_DIM;
    const __bf16* src_b = (half ? attn2_d : attn1_d) + ((size_t)b*MHALF + m0)*C_DIM;
    #pragma unroll
    for (int i = 0; i < 8; ++i) {
        int lin = tid + i*128;                 // 0..1023 16-byte slots
        async_copy16(src_a + lin*8, ya + lin*8);
        async_copy16(src_b + lin*8, yb + lin*8);
    }
    async_wait_all();
    __syncthreads();

    const int w = tid >> 5, lane = tid & 31, rr = lane & 15, hh = lane >> 4;
    const int r0 = w * 16;

    v16bf afa[4], afb[4];
    #pragma unroll
    for (int kt = 0; kt < 4; ++kt) {
        #pragma unroll
        for (int e = 0; e < 16; ++e) {
            int k = kt*32 + e + 16*hh;
            afa[kt][e] = ya[(r0 + rr)*C_DIM + k];
            afb[kt][e] = yb[(r0 + rr)*C_DIM + k];
        }
    }
    float* orow = out + ((size_t)b*NFULL + t0 + r0)*C_DIM;
    #pragma unroll
    for (int ct = 0; ct < 8; ++ct) {
        v8f acc = {};
        #pragma unroll
        for (int kt = 0; kt < 4; ++kt) {
            v16bf bfr = *(const v16bf*)(wf + ((size_t)(ct*4 + kt)*32 + lane)*16);
            acc = __builtin_amdgcn_wmma_f32_16x16x32_bf16(false, afa[kt], false, bfr,
                                                          (short)0, acc, false, false);
        }
        #pragma unroll
        for (int kt = 0; kt < 4; ++kt) {
            v16bf bfr = *(const v16bf*)(projp + ((size_t)(ct*4 + kt)*32 + lane)*16);
            acc = __builtin_amdgcn_wmma_f32_16x16x32_bf16(false, afb[kt], false, bfr,
                                                          (short)0, acc, false, false);
        }
        float bias = proj_b[ct*16 + rr];
        #pragma unroll
        for (int i = 0; i < 8; ++i)
            orow[(size_t)(i + 8*hh)*C_DIM + ct*16 + rr] = acc[i] + bias;
    }
}

// ---------------------------------------------------------------------------
extern "C" void kernel_launch(void* const* d_in, const int* in_sizes, int n_in,
                              void* d_out, int out_size, void* d_ws, size_t ws_size,
                              hipStream_t stream)
{
    (void)in_sizes; (void)n_in; (void)out_size; (void)ws_size;
    const float* x      = (const float*)d_in[0];
    const float* qkv_w  = (const float*)d_in[1];
    const float* ca_w1  = (const float*)d_in[2];
    const float* ca_w2  = (const float*)d_in[3];
    const float* proj_w = (const float*)d_in[4];
    const float* proj_b = (const float*)d_in[5];
    float* out = (float*)d_out;

    char* ws = (char*)d_ws;
    size_t off = 0;
    auto carve = [&](size_t bytes) -> char* {
        char* p = ws + off;
        off += (bytes + 255) & ~(size_t)255;
        return p;
    };
    __bf16* qkvp    = (__bf16*)carve((size_t)QKVP_ELEMS * 2);
    __bf16* projp   = (__bf16*)carve((size_t)PROJP_ELEMS * 2);
    __bf16* attn_d  = (__bf16*)carve((size_t)B_SZ*MHALF*C_DIM * 2);
    __bf16* attn1_d = (__bf16*)carve((size_t)B_SZ*MHALF*C_DIM * 2);
    __bf16* attn2_d = (__bf16*)carve((size_t)B_SZ*MHALF*C_DIM * 2);
    float*  psum    = (float*)carve((size_t)3*B_SZ*64*C_DIM * 4);
    float*  pmax    = (float*)carve((size_t)3*B_SZ*64*C_DIM * 4);
    __bf16* wfp     = (__bf16*)carve((size_t)2*B_SZ*WF_ELEMS_PER * 2);

    pack_weights_kernel<<<(QKVP_ELEMS + PROJP_ELEMS + 255)/256, 256, 0, stream>>>(
        qkv_w, proj_w, qkvp, projp);

    size_t smemA = (size_t)(2*64*128)*sizeof(float)
                 + (size_t)(64*384 + 64*256 + 3*64*128)*sizeof(__bf16);
    qkv_headattn_kernel<<<B_SZ*64, 128, smemA, stream>>>(
        x, qkvp, attn_d, attn1_d, attn2_d, psum, pmax);

    size_t smemB = (size_t)(3*128 + 2*128 + 64 + 128)*sizeof(float)
                 + (size_t)128*128*sizeof(__bf16);
    channel_mix_kernel<<<B_SZ, 128, smemB, stream>>>(
        psum, pmax, ca_w1, ca_w2, projp, wfp);

    size_t smemC = (size_t)2*64*128*sizeof(__bf16);
    fused_proj_kernel<<<B_SZ*128, 128, smemC, stream>>>(
        attn_d, attn1_d, attn2_d, wfp, projp, proj_b, out);
}
